// LinearAttention_68865505624102
// MI455X (gfx1250) — matrix operand order repaired and checked by reading, not compiled, not run on hardware
//
#include <hip/hip_runtime.h>
#include <hip/hip_bf16.h>

// ---------------------------------------------------------------------------
// CDNA5 (gfx1250) WMMA types
// ---------------------------------------------------------------------------
typedef __attribute__((ext_vector_type(16))) __bf16 v16bf;
typedef __attribute__((ext_vector_type(8)))  __bf16 v8bf;
typedef __attribute__((ext_vector_type(2)))  __bf16 v2bf;
typedef __attribute__((ext_vector_type(8)))  float  v8f;
typedef __attribute__((ext_vector_type(4)))  int    v4i;

#define WMMA_BF16(a, b, c) \
  __builtin_amdgcn_wmma_f32_16x16x32_bf16(false, (a), false, (b), (short)0, (c), false, false)

// ---------------------------------------------------------------------------
// Async global->LDS copy (GLOBAL_LOAD_ASYNC_TO_LDS_B128), guarded fallback.
// Probe-confirmed: builtin takes (v4i* src, v4i* dst, imm offset, imm cpol).
// ---------------------------------------------------------------------------
#if defined(__AMDGCN__) && __has_builtin(__builtin_amdgcn_global_load_async_to_lds_b128)
#define HAVE_ASYNC_LDS 1
#else
#define HAVE_ASYNC_LDS 0
#endif

__device__ __forceinline__ void async_copy16(const __bf16* __restrict__ src, __bf16* dst) {
#if HAVE_ASYNC_LDS
  __builtin_amdgcn_global_load_async_to_lds_b128((v4i*)src, (v4i*)dst, 0, 0);
#else
  *(uint4*)dst = *(const uint4*)src;
#endif
}

__device__ __forceinline__ void async_join() {
#if HAVE_ASYNC_LDS
#if __has_builtin(__builtin_amdgcn_s_wait_asynccnt)
  __builtin_amdgcn_s_wait_asynccnt(0);
#else
  asm volatile("s_wait_asynccnt 0" ::: "memory");
#endif
#endif
}

// ---------------------------------------------------------------------------
// Fragment helpers (wave32; A and B both use contiguous k = 16*(lane/16) + s)
// ---------------------------------------------------------------------------
// A: 16x32 bf16 tile, row-major row stride ldm.
__device__ __forceinline__ v16bf load_frag_a(const __bf16* base, int ldm) {
  int lane = threadIdx.x & 31;
  int row  = lane & 15;
  int h    = lane >> 4;
  const __bf16* p = base + (size_t)row * ldm + h * 16;
  union { v16bf v; v8bf half[2]; } u;
  u.half[0] = *(const v8bf*)(p);
  u.half[1] = *(const v8bf*)(p + 8);
  return u.v;
}

// B: 32(K)x16(N) tile stored transposed [n][k], n-stride ldn (LDS or global).
__device__ __forceinline__ v16bf load_frag_b(const __bf16* base, int ldn) {
  int lane = threadIdx.x & 31;
  int n    = lane & 15;
  int kh   = lane >> 4;
  return *(const v16bf*)(base + (size_t)n * ldn + kh * 16);
}

// C/D 16x16 f32 tile -> bf16, row-major output (scattered b16 stores).
__device__ __forceinline__ void store_tile_bf16(__bf16* out, int ld, v8f c) {
  int lane = threadIdx.x & 31;
  int n = lane & 15, mh = lane >> 4;
#pragma unroll
  for (int r = 0; r < 8; ++r)
    out[(size_t)(mh * 8 + r) * ld + n] = (__bf16)c[r];
}

// C/D tile -> bf16, TRANSPOSED output [n][m]: one contiguous 16B store/lane.
__device__ __forceinline__ void store_tile_bf16_T(__bf16* outT, int ldT, v8f c) {
  int lane = threadIdx.x & 31;
  int n = lane & 15, mh = lane >> 4;
  v8bf o;
#pragma unroll
  for (int r = 0; r < 8; ++r) o[r] = (__bf16)c[r];
  *(v8bf*)(outT + (size_t)n * ldT + mh * 8) = o;
}

__device__ __forceinline__ void atomic_tile_f32(float* out, int ld, v8f c) {
  int lane = threadIdx.x & 31;
  int n = lane & 15, mh = lane >> 4;
#pragma unroll
  for (int r = 0; r < 8; ++r)
    atomicAdd(&out[(size_t)(mh * 8 + r) * ld + n], c[r]);
}

// ---------------------------------------------------------------------------
// Kernel 0: convert the four 512x512 f32 weight matrices to bf16 (once/call).
// ---------------------------------------------------------------------------
__global__ __launch_bounds__(256) void cvt_w(const float* __restrict__ W0,
                                             const float* __restrict__ W1,
                                             const float* __restrict__ W2,
                                             const float* __restrict__ W3,
                                             __bf16* __restrict__ out) {
  const float* src = blockIdx.y == 0 ? W0 : blockIdx.y == 1 ? W1
                   : blockIdx.y == 2 ? W2 : W3;
  __bf16* dst = out + (size_t)blockIdx.y * 512 * 512;
  int i = (blockIdx.x * 256 + threadIdx.x) * 8;
  float4 a = *(const float4*)(src + i);
  float4 b = *(const float4*)(src + i + 4);
  v8bf o = {(__bf16)a.x, (__bf16)a.y, (__bf16)a.z, (__bf16)a.w,
            (__bf16)b.x, (__bf16)b.y, (__bf16)b.z, (__bf16)b.w};
  *(v8bf*)(dst + i) = o;
}

// ---------------------------------------------------------------------------
// Kernel 1: projection GEMM.  Out[b] = W(bf16 [M][K]) @ In[b](f32 [K][N]) -> bf16.
// TOUT=false: Out row-major [M][N].  TOUT=true: Out transposed [N][M].
// Block tile 128x128, K-step 32, 8 waves (2x4), wave tile 64x32.
// A-tile staged by async DMA from bf16 weights; B-tile converted f32->bf16
// with b32-packed transposed stores into padded LDS.
// ---------------------------------------------------------------------------
#define PBM 128
#define PBN 128
#define PBK 32
#define LDA_P 40   // padded bf16 row stride for A tiles (bank-clean b128 reads)
#define LDB_P 34   // padded bf16 n-stride for B tiles

template <bool TOUT>
__global__ __launch_bounds__(256) void proj_gemm(const __bf16* __restrict__ W,
                                                 const float* __restrict__ In,
                                                 __bf16* __restrict__ Out,
                                                 int M, int K, int N) {
  __shared__ __align__(32) __bf16 ldsA[PBM * LDA_P];
  __shared__ __align__(32) __bf16 ldsB[PBN * LDB_P];
  const int b  = blockIdx.z;
  const int m0 = blockIdx.y * PBM;
  const int n0 = blockIdx.x * PBN;
  const float* inB  = In  + (size_t)b * K * N;
  __bf16*      outB = Out + (size_t)b * M * N;
  const int tid  = threadIdx.x;
  const int wave = tid >> 5;
  const int wm   = wave & 1;   // 0..1 -> 64 rows
  const int wn   = wave >> 1;  // 0..3 -> 32 cols

  v8f acc[4][2] = {};

  for (int k0 = 0; k0 < K; k0 += PBK) {
    // A: 128x32 bf16 via async DMA, 512 16B-chunks / 256 threads.
#pragma unroll
    for (int i = 0; i < 2; ++i) {
      int idx = tid + i * 256;
      int row = idx >> 2, c8 = (idx & 3) * 8;
      async_copy16(W + (size_t)(m0 + row) * K + k0 + c8, ldsA + row * LDA_P + c8);
    }
    // B: 32x128 f32 -> bf16 transposed [n][k]; 2x2 patches, b32 LDS stores.
#pragma unroll
    for (int i = 0; i < 4; ++i) {
      int idx = tid + i * 256;              // 1024 patches
      int n = (idx & 63) * 2;
      int k = (idx >> 6) * 2;
      const float* s0 = inB + (size_t)(k0 + k) * N + n0 + n;
      float2 r0 = *(const float2*)(s0);
      float2 r1 = *(const float2*)(s0 + N);
      *(v2bf*)(ldsB + (n + 0) * LDB_P + k) = v2bf{(__bf16)r0.x, (__bf16)r1.x};
      *(v2bf*)(ldsB + (n + 1) * LDB_P + k) = v2bf{(__bf16)r0.y, (__bf16)r1.y};
    }
    async_join();
    __syncthreads();

    v16bf bfr[2];
#pragma unroll
    for (int j = 0; j < 2; ++j)
      bfr[j] = load_frag_b(ldsB + (wn * 32 + j * 16) * LDB_P, LDB_P);
#pragma unroll
    for (int i = 0; i < 4; ++i) {
      v16bf afr = load_frag_a(ldsA + (wm * 64 + i * 16) * LDA_P, LDA_P);
#pragma unroll
      for (int j = 0; j < 2; ++j)
        acc[i][j] = WMMA_BF16(afr, bfr[j], acc[i][j]);
    }
    __syncthreads();
  }

#pragma unroll
  for (int i = 0; i < 4; ++i)
#pragma unroll
    for (int j = 0; j < 2; ++j) {
      int mb = m0 + wm * 64 + i * 16;
      int nb = n0 + wn * 32 + j * 16;
      if (TOUT)   // transposed [N][M]: contiguous 16B stores per lane
        store_tile_bf16_T(outB + (size_t)nb * M + mb, M, acc[i][j]);
      else
        store_tile_bf16(outB + (size_t)mb * N + nb, N, acc[i][j]);
    }
}

// ---------------------------------------------------------------------------
// Kernel 2: masked softmax over key-time axis, in-place on bf16 k buffer.
// ---------------------------------------------------------------------------
__global__ __launch_bounds__(256) void softmax_k(__bf16* __restrict__ Kbuf,
                                                 const unsigned char* __restrict__ mask,
                                                 int C, int T) {
  const int row = blockIdx.x;           // [0, B*C)
  const int b   = row / C;
  __bf16* kr = Kbuf + (size_t)row * T;
  const unsigned char* mr = mask + (size_t)b * T;
  __shared__ float red[256];
  const int tid = threadIdx.x;

  float vals[16];
  float lmax = -3.4e38f;
#pragma unroll
  for (int i = 0; i < 16; ++i) {
    int t = tid + i * 256;
    float v = (float)kr[t];
    if (mr[t]) v = -10000.0f;
    v += 1e-4f;
    vals[i] = v;
    lmax = fmaxf(lmax, v);
  }
  red[tid] = lmax; __syncthreads();
  for (int s = 128; s > 0; s >>= 1) {
    if (tid < s) red[tid] = fmaxf(red[tid], red[tid + s]);
    __syncthreads();
  }
  const float m = red[0]; __syncthreads();

  float lsum = 0.f;
#pragma unroll
  for (int i = 0; i < 16; ++i) { vals[i] = __expf(vals[i] - m); lsum += vals[i]; }
  red[tid] = lsum; __syncthreads();
  for (int s = 128; s > 0; s >>= 1) {
    if (tid < s) red[tid] += red[tid + s];
    __syncthreads();
  }
  const float inv = 1.0f / red[0];
#pragma unroll
  for (int i = 0; i < 16; ++i)
    kr[tid + i * 256] = (__bf16)(vals[i] * inv);
}

// ---------------------------------------------------------------------------
// Kernel 3: ctx[b,h] (f32 64x64) += Ksoft[b,h] @ V[b,h]^T, split-K atomics.
// Both operands t-contiguous -> fragments straight from global memory.
// ---------------------------------------------------------------------------
__global__ __launch_bounds__(256) void ctx_gemm(const __bf16* __restrict__ Ks,
                                                const __bf16* __restrict__ V,
                                                float* __restrict__ Ctx,
                                                int T, int kchunk) {
  const int bh = blockIdx.x;            // [0, B*H)
  const int k0beg = blockIdx.y * kchunk;
  const int k0end = k0beg + kchunk;
  const __bf16* A  = Ks + (size_t)bh * 64 * T;
  const __bf16* Bm = V  + (size_t)bh * 64 * T;
  float* C = Ctx + (size_t)bh * 64 * 64;
  const int wave = threadIdx.x >> 5;
  const int mi = wave & 3;
  const int n0 = (wave >> 2) * 32;

  v8f acc[2] = {};
  for (int k0 = k0beg; k0 < k0end; k0 += 32) {
    __builtin_prefetch(A + (size_t)(mi * 16) * T + k0 + 256, 0, 1);
    v16bf af = load_frag_a(A + (size_t)(mi * 16) * T + k0, T);
#pragma unroll
    for (int j = 0; j < 2; ++j) {
      v16bf bf = load_frag_b(Bm + (size_t)(n0 + j * 16) * T + k0, T);
      acc[j] = WMMA_BF16(af, bf, acc[j]);
    }
  }
#pragma unroll
  for (int j = 0; j < 2; ++j)
    atomic_tile_f32(C + (size_t)(mi * 16) * 64 + n0 + j * 16, 64, acc[j]);
}

// ---------------------------------------------------------------------------
// Kernel 4: attnT[b, t, h*64+dv] = sum_dk ctx[b,h,dk,dv] * qT[b, t, h*64+dk]
// qT is [b][t][c] -> B fragments are K-contiguous, loaded straight from global.
// Output written transposed -> contiguous 16B stores.
// ---------------------------------------------------------------------------
#define LDC_P 72   // padded stride for ctx^T in LDS
__global__ __launch_bounds__(256) void av_gemm(const float* __restrict__ Ctx,
                                               const __bf16* __restrict__ Qt,
                                               __bf16* __restrict__ AttnT, int T) {
  __shared__ __align__(32) __bf16 ldsA[64 * LDC_P];   // ctx^T [dv][dk]
  const int t0 = blockIdx.x * 128;
  const int h  = blockIdx.y;
  const int b  = blockIdx.z;
  const float*  ctx  = Ctx   + ((size_t)(b * 8 + h)) * 64 * 64;
  const __bf16* qt   = Qt    + (size_t)b * T * 512 + h * 64;
  __bf16*       outT = AttnT + (size_t)b * T * 512 + h * 64;
  const int tid = threadIdx.x;

#pragma unroll
  for (int i = 0; i < 16; ++i) {              // 4096 elems / 256 thr
    int idx = tid + i * 256;
    int dk = idx >> 6, dv = idx & 63;
    ldsA[dv * LDC_P + dk] = (__bf16)ctx[dk * 64 + dv];
  }
  __syncthreads();

  const int wave = tid >> 5;
  const int n0 = wave * 16;
  v8f acc[4] = {};
#pragma unroll
  for (int k0 = 0; k0 < 64; k0 += 32) {
    v16bf bf = load_frag_b(qt + (size_t)(t0 + n0) * 512 + k0, 512);
#pragma unroll
    for (int i = 0; i < 4; ++i) {
      v16bf af = load_frag_a(ldsA + (i * 16) * LDC_P + k0, LDC_P);
      acc[i] = WMMA_BF16(af, bf, acc[i]);
    }
  }
#pragma unroll
  for (int i = 0; i < 4; ++i)
    store_tile_bf16_T(outT + (size_t)(t0 + n0) * 512 + i * 16, 512, acc[i]);
}

// ---------------------------------------------------------------------------
// Kernel 5: P = Wp @ attn + bp;  out = LN(scale*P + residual)*gamma + beta
// Full-C (M=512) x 32-column block tile; B fragments straight from global
// (attnT is K-contiguous); Wp A-tiles via async DMA; LDS reused for the
// padded f32 LayerNorm column buffer.
// ---------------------------------------------------------------------------
__global__ __launch_bounds__(256) void final_kernel(const __bf16* __restrict__ Wp,
                                                    const __bf16* __restrict__ AttnT,
                                                    const float* __restrict__ bp,
                                                    const float* __restrict__ scale,
                                                    const float* __restrict__ residual,
                                                    const float* __restrict__ gamma,
                                                    const float* __restrict__ beta,
                                                    float* __restrict__ Out, int T) {
  __shared__ __align__(32) char smem[512 * 33 * 4];   // 66 KiB, phase-reused
  __bf16* ldsA = (__bf16*)smem;                        // 512 x LDA_P bf16 (GEMM)
  float*  ldsP = (float*)smem;                         // 512 x 33 f32 (LN)
  __shared__ float ldsMean[32], ldsInv[32];

  const int t0 = blockIdx.x * 32;
  const int b  = blockIdx.y;
  const __bf16* attnT = AttnT    + (size_t)b * T * 512;
  const float*  res   = residual + (size_t)b * 512 * T;
  float*        out   = Out      + (size_t)b * 512 * T;
  const int tid  = threadIdx.x;
  const int wave = tid >> 5;
  const int lane = tid & 31;

  v8f acc[4][2] = {};
  for (int k0 = 0; k0 < 512; k0 += 32) {
    // A: Wp 512x32 bf16 via async DMA: 2048 16B-chunks / 256 threads.
#pragma unroll
    for (int i = 0; i < 8; ++i) {
      int idx = tid + i * 256;
      int row = idx >> 2, c8 = (idx & 3) * 8;
      async_copy16(Wp + (size_t)row * 512 + k0 + c8, ldsA + row * LDA_P + c8);
    }
    async_join();
    __syncthreads();

    v16bf bf0 = load_frag_b(attnT + (size_t)(t0 +  0) * 512 + k0, 512);
    v16bf bf1 = load_frag_b(attnT + (size_t)(t0 + 16) * 512 + k0, 512);
#pragma unroll
    for (int i = 0; i < 4; ++i) {
      v16bf af = load_frag_a(ldsA + (wave * 64 + i * 16) * LDA_P, LDA_P);
      acc[i][0] = WMMA_BF16(af, bf0, acc[i][0]);
      acc[i][1] = WMMA_BF16(af, bf1, acc[i][1]);
    }
    __syncthreads();
  }

  // scale*(P + bp) + residual -> padded LDS column buffer
  const float sc = scale[0];
  const int nloc = lane & 15, mh = lane >> 4;
#pragma unroll
  for (int i = 0; i < 4; ++i)
#pragma unroll
    for (int j = 0; j < 2; ++j) {
      int cbase = wave * 64 + i * 16 + mh * 8;
      int n = j * 16 + nloc;
#pragma unroll
      for (int r = 0; r < 8; ++r) {
        int c = cbase + r;
        ldsP[c * 33 + n] = sc * (acc[i][j][r] + bp[c]) + res[(size_t)c * T + t0 + n];
      }
    }
  __syncthreads();

  // LayerNorm stats: 8 waves x 4 columns; lane l covers c = r*32 + l
#pragma unroll
  for (int cc = 0; cc < 4; ++cc) {
    int n = wave * 4 + cc;
    float s1 = 0.f, s2 = 0.f;
#pragma unroll
    for (int r = 0; r < 16; ++r) {
      float v = ldsP[(r * 32 + lane) * 33 + n];
      s1 += v; s2 += v * v;
    }
#pragma unroll
    for (int off = 16; off > 0; off >>= 1) {
      s1 += __shfl_xor(s1, off, 32);
      s2 += __shfl_xor(s2, off, 32);
    }
    if (lane == 0) {
      float mean = s1 * (1.0f / 512.0f);
      float var  = s2 * (1.0f / 512.0f) - mean * mean;
      ldsMean[n] = mean;
      ldsInv[n]  = rsqrtf(var + 1e-5f);
    }
  }
  __syncthreads();

  // Coalesced normalize + affine + store
  const int n  = tid & 31;
  const int c0 = tid >> 5;
  const float mean = ldsMean[n], inv = ldsInv[n];
#pragma unroll
  for (int i = 0; i < 64; ++i) {
    int c = c0 + i * 8;
    float v = (ldsP[c * 33 + n] - mean) * inv;
    out[(size_t)c * T + t0 + n] = v * gamma[c] + beta[c];
  }
}

// ---------------------------------------------------------------------------
// Host launch
// ---------------------------------------------------------------------------
extern "C" void kernel_launch(void* const* d_in, const int* in_sizes, int n_in,
                              void* d_out, int out_size, void* d_ws, size_t ws_size,
                              hipStream_t stream) {
  const float* queries = (const float*)d_in[0];
  const float* keys    = (const float*)d_in[1];
  const float* values  = (const float*)d_in[2];
  const unsigned char* kpm = (const unsigned char*)d_in[3];
  const float* Wq = (const float*)d_in[4];
  const float* Wk = (const float*)d_in[5];
  const float* Wv = (const float*)d_in[6];
  const float* Wp = (const float*)d_in[7];
  const float* bp = (const float*)d_in[8];
  const float* scale = (const float*)d_in[9];
  const float* gamma = (const float*)d_in[10];
  const float* beta  = (const float*)d_in[11];
  float* out = (float*)d_out;

  const int B = 8, C = 512, T = 4096, H = 8;
  const size_t tens = (size_t)B * C * T * sizeof(__bf16);             // 32 MiB
  const size_t ctxBytes = (size_t)B * H * 64 * 64 * sizeof(float);    // 1 MiB
  char* ws = (char*)d_ws;
  __bf16* q_ws    = (__bf16*)(ws);                    // [b][t][c] transposed
  __bf16* k_ws    = (__bf16*)(ws + tens);             // [b][c][t]
  __bf16* v_ws    = (__bf16*)(ws + 2 * tens);         // [b][c][t]
  float*  ctx_ws  = (float*)(ws + 3 * tens);          // [b][h][64][64]
  __bf16* attn_ws = (__bf16*)(ws + 3 * tens + ctxBytes);  // [b][t][c] transposed
  __bf16* wbf     = (__bf16*)(ws + 4 * tens + ctxBytes);  // 4 x 512x512 bf16
  __bf16* wq_bf = wbf;
  __bf16* wk_bf = wbf + (size_t)C * C;
  __bf16* wv_bf = wbf + (size_t)2 * C * C;
  __bf16* wp_bf = wbf + (size_t)3 * C * C;

  cvt_w<<<dim3(C * C / (256 * 8), 4), 256, 0, stream>>>(Wq, Wk, Wv, Wp, wbf);

  dim3 gP(T / 128, C / 128, B);
  proj_gemm<true ><<<gP, 256, 0, stream>>>(wq_bf, queries, q_ws, C, C, T);
  proj_gemm<false><<<gP, 256, 0, stream>>>(wk_bf, keys,    k_ws, C, C, T);
  proj_gemm<false><<<gP, 256, 0, stream>>>(wv_bf, values,  v_ws, C, C, T);

  softmax_k<<<B * C, 256, 0, stream>>>(k_ws, kpm, C, T);

  (void)hipMemsetAsync(ctx_ws, 0, ctxBytes, stream);
  ctx_gemm<<<dim3(B * H, 4), 256, 0, stream>>>(k_ws, v_ws, ctx_ws, T, T / 4);

  av_gemm<<<dim3(T / 128, H, B), 256, 0, stream>>>(ctx_ws, q_ws, attn_ws, T);

  final_kernel<<<dim3(T / 32, B), 256, 0, stream>>>(wp_bf, attn_ws, bp, scale,
                                                    queries, gamma, beta, out, T);
}
